// PSA_Block_47931835023508
// MI455X (gfx1250) — compile-verified, matrix operand order checked by reading, not run on hardware
//
#include <hip/hip_runtime.h>

typedef __attribute__((ext_vector_type(16))) __bf16 bf16x16;
typedef __attribute__((ext_vector_type(8)))  __bf16 bf16x8;
typedef __attribute__((ext_vector_type(8)))  float  f32x8;
typedef __attribute__((ext_vector_type(2)))  float  f32x2;

#define H_   256
#define W_   256
#define C_   64
#define NH_  4
#define HID_ 128
#define NTOK (H_*W_)
#define SCALE_ 0.25f

#if defined(__has_builtin)
#if __has_builtin(__builtin_amdgcn_wmma_f32_16x16x4_f32)
#define HAVE_WMMA_F32 1
#endif
#endif
#ifndef HAVE_WMMA_F32
#warning "gfx1250: __builtin_amdgcn_wmma_f32_16x16x4_f32 NOT declared; QK uses bf16 fallback"
#endif

__device__ __forceinline__ float gelu_exact(float v) {
  return 0.5f * v * (1.0f + erff(v * 0.70710678118654752f));
}

// Fragment loaders.  CDNA5 16x16x32 bf16 A/B k-map: element e -> k = (e&7)+8g+16(e>>3)
// => per lane two contiguous 8-element (16B) chunks at kBase+8g and kBase+8g+16.
__device__ __forceinline__ bf16x16 ld_frag_bf16(const __bf16* p) {
  bf16x8 lo = *(const bf16x8*)p;
  bf16x8 hi = *(const bf16x8*)(p + 16);
  return __builtin_shufflevector(lo, hi, 0,1,2,3,4,5,6,7,8,9,10,11,12,13,14,15);
}
__device__ __forceinline__ bf16x16 ld_frag_f32cvt(const float* p) {
  f32x8 lo = *(const f32x8*)p;
  f32x8 hi = *(const f32x8*)(p + 16);
  bf16x16 r;
#pragma unroll
  for (int e = 0; e < 8; ++e) { r[e] = (__bf16)lo[e]; r[e+8] = (__bf16)hi[e]; }
  return r;
}

// Async global->LDS copy (ASYNCcnt); LDS offset = low 32 bits of flat shared ptr.
__device__ __forceinline__ unsigned lds_off(const void* p) {
  return (unsigned)(unsigned long long)p;
}
__device__ __forceinline__ void async_b128(unsigned lds, const void* g) {
  asm volatile("global_load_async_to_lds_b128 %0, %1, off" :: "v"(lds), "v"(g) : "memory");
}
__device__ __forceinline__ void async_wait0() {
  asm volatile("s_wait_asynccnt 0x0" ::: "memory");
}

// ---------------------------------------------------------------------------
// K1: LN1 + offset-predictor MLP. One thread = one token.
// Writes xn (bf16) and gather index fidx.
// ---------------------------------------------------------------------------
__global__ void __launch_bounds__(256)
k_ln1_pred(const float* __restrict__ x,
           const float* __restrict__ g1, const float* __restrict__ b1,
           const float* __restrict__ pw1, const float* __restrict__ pb1,
           const float* __restrict__ pw2, const float* __restrict__ pb2,
           __bf16* __restrict__ xn, int* __restrict__ fidx) {
  __shared__ float sw1[64*32];
  __shared__ float sw2[32*2];
  __shared__ float sb1[32], sb2[2], sg[64], sbv[64];
  int tid = threadIdx.x;
  for (int i = tid; i < 64*32; i += 256) sw1[i] = pw1[i];
  if (tid < 64) { sw2[tid] = pw2[tid]; sg[tid] = g1[tid]; sbv[tid] = b1[tid]; }
  if (tid < 32) sb1[tid] = pb1[tid];
  if (tid < 2)  sb2[tid] = pb2[tid];
  __syncthreads();

  long tok = (long)blockIdx.x * 256 + tid;
  const float* xr = x + tok * C_;
  float v[C_];
  float mean = 0.f;
#pragma unroll
  for (int i = 0; i < C_; ++i) { v[i] = xr[i]; mean += v[i]; }
  mean *= (1.f/64.f);
  float var = 0.f;
#pragma unroll
  for (int i = 0; i < C_; ++i) { float d = v[i]-mean; var += d*d; }
  float rs = rsqrtf(var*(1.f/64.f) + 1e-5f);
  __bf16* xo = xn + tok * C_;
#pragma unroll
  for (int i = 0; i < C_; ++i) { v[i] = (v[i]-mean)*rs*sg[i] + sbv[i]; xo[i] = (__bf16)v[i]; }

  float o0 = sb2[0], o1 = sb2[1];
  for (int j = 0; j < 32; ++j) {
    float t = sb1[j];
#pragma unroll
    for (int i = 0; i < 64; ++i) t += v[i]*sw1[i*32+j];
    t = t >= 0.f ? t : 0.2f*t;
    o0 += t * sw2[j*2+0];
    o1 += t * sw2[j*2+1];
  }
  int t32 = (int)(tok & (NTOK-1));
  int gy = t32 >> 8, gx = t32 & 255;
  int ny = gy + (int)rintf(o0); ny = min(max(ny,0), H_-1);
  int nx = gx + (int)rintf(o1); nx = min(max(nx,0), W_-1);
  fidx[tok] = ny * W_ + nx;
}

// ---------------------------------------------------------------------------
// K2: async-gather + window attention + proj + residual. One wave = one window.
// ---------------------------------------------------------------------------
__global__ void __launch_bounds__(32)
k_attn(const float* __restrict__ x, const __bf16* __restrict__ xn,
       const int* __restrict__ fidx,
       const float* __restrict__ qw, const float* __restrict__ qb,
       const float* __restrict__ kvw, const float* __restrict__ kvb,
       const float* __restrict__ pw,  const float* __restrict__ pb,
       const float* __restrict__ rpb,
       float* __restrict__ xmid) {
  __shared__ __align__(32) char smem[61840];
  float*  kvk  = (float*)(smem);                 // [64 tok][32 ch]  K, token-major
  float*  vT   = (float*)(smem + 8192);          // [32 ch][64 tok]  V, transposed
  float*  qs   = (float*)(smem + 16384);         // [64][32]
  float*  sc   = (float*)(smem + 24576);         // [64][64] scores/probs (f32)
  __bf16* xw   = (__bf16*)(smem + 24576);        // [64][64] gathered rows (aliases sc)
  __bf16* ao   = (__bf16*)(smem + 40960);        // [64][32] attn out
  __bf16* wqT  = (__bf16*)(smem + 45056);        // [32 n][64 k]
  __bf16* wpT  = (__bf16*)(smem + 45056);        // [64 n][32 k] (aliases wqT)
  __bf16* wkvT = (__bf16*)(smem + 49152);        // [64 n][64 k]
  float*  rpbs = (float*)(smem + 57344);         // [225][4]
  float*  qbs  = (float*)(smem + 60944);
  float*  kvbs = (float*)(smem + 61072);
  float*  pbs  = (float*)(smem + 61328);
  int*    fx   = (int*)(smem + 61584);

  int lane = threadIdx.x;
  int wid  = blockIdx.x;
  int batch = wid >> 10;
  int widx  = wid & 1023;
  int wy = widx >> 5, wx = widx & 31;
  long bbase = (long)batch * NTOK;

  for (int i = lane; i < 64; i += 32) {
    int py = i >> 3, px = i & 7;
    fx[i] = fidx[bbase + (long)((wy*8+py)*W_ + (wx*8+px))];
    kvbs[i] = kvb[i];
    pbs[i]  = pb[i];
  }
  // transposed bf16 weight tiles
  for (int i = lane; i < 32*64; i += 32) { int n = i >> 6, k = i & 63; wqT[i]  = (__bf16)qw[k*32+n]; }
  for (int i = lane; i < 64*64; i += 32) { int n = i >> 6, k = i & 63; wkvT[i] = (__bf16)kvw[k*64+n]; }
  for (int i = lane; i < 225*4; i += 32) rpbs[i] = rpb[i];
  if (lane < 32) qbs[lane] = qb[lane];
  __syncthreads();

  // async gather: 64 rows of 128B each (8 x b128 chunks) into xw
  unsigned xwb = lds_off(xw);
  for (int i = lane; i < 64*8; i += 32) {
    int r = i >> 3, c8 = (i & 7) << 3;
    async_b128(xwb + (unsigned)(r*64 + c8)*2u, xn + (bbase + fx[r]) * C_ + c8);
  }
  async_wait0();
  __syncthreads();

  int lm = lane & 15, lg = lane >> 4;

  // --- QKV: [64x64] @ [64x96] (bf16 WMMA) ---
  for (int mt = 0; mt < 4; ++mt)
    for (int nt = 0; nt < 6; ++nt) {
      f32x8 c = {};
#pragma unroll
      for (int kt = 0; kt < 2; ++kt) {
        const __bf16* bp = (nt < 2) ? (wqT  + (nt*16+lm)*64)
                                    : (wkvT + ((nt-2)*16+lm)*64);
        bf16x16 a = ld_frag_bf16(xw + (mt*16 + lm)*64 + kt*32 + 8*lg);
        bf16x16 b = ld_frag_bf16(bp + kt*32 + 8*lg);
        c = __builtin_amdgcn_wmma_f32_16x16x32_bf16(false, a, false, b, (short)0, c, false, false);
      }
#pragma unroll
      for (int r = 0; r < 8; ++r) {
        int m = mt*16 + r + (lg << 3);
        if (nt < 2)      { int n = nt*16 + lm;     qs [m*32 + n] = c[r] + qbs[n]; }
        else if (nt < 4) { int n = (nt-2)*16 + lm; kvk[m*32 + n] = c[r] + kvbs[n]; }
        else             { int n = (nt-4)*16 + lm; vT [n*64 + m] = c[r] + kvbs[32+n]; }
      }
    }
  __syncthreads();

  for (int h = 0; h < NH_; ++h) {
    // --- scores = q_h @ k_h^T * SCALE + bias ---
    for (int mt = 0; mt < 4; ++mt)
      for (int nt = 0; nt < 4; ++nt) {
        f32x8 c = {};
#if defined(HAVE_WMMA_F32)
#pragma unroll
        for (int kt = 0; kt < 2; ++kt) {
          f32x2 a = *(const f32x2*)(qs  + (mt*16+lm)*32 + h*8 + kt*4 + 2*lg);
          f32x2 b = *(const f32x2*)(kvk + (nt*16+lm)*32 + h*8 + kt*4 + 2*lg);
          c = __builtin_amdgcn_wmma_f32_16x16x4_f32(false, a, false, b, (short)0, c, false, false);
        }
#else
        {
          bf16x16 a, b;
#pragma unroll
          for (int e = 0; e < 16; ++e) {
            int kk = (e & 7) + (lg << 3) + ((e >> 3) << 4);   // pad K 8->32
            a[e] = (kk < 8) ? (__bf16)qs [(mt*16+lm)*32 + h*8 + kk] : (__bf16)0.f;
            b[e] = (kk < 8) ? (__bf16)kvk[(nt*16+lm)*32 + h*8 + kk] : (__bf16)0.f;
          }
          c = __builtin_amdgcn_wmma_f32_16x16x32_bf16(false, a, false, b, (short)0, c, false, false);
        }
#endif
#pragma unroll
        for (int r = 0; r < 8; ++r) {
          int qm = mt*16 + r + (lg << 3);
          int kn = nt*16 + lm;
          int dy = (qm >> 3) - (kn >> 3) + 7;
          int dx = (qm & 7)  - (kn & 7)  + 7;
          sc[qm*64 + kn] = c[r]*SCALE_ + rpbs[(dy*15 + dx)*4 + h];
        }
      }
    __syncthreads();

    // --- softmax (2 rows per lane) ---
    for (int rr = 0; rr < 2; ++rr) {
      float* srow = sc + (lane*2 + rr)*64;
      float mx = srow[0];
      for (int j = 1; j < 64; ++j) mx = fmaxf(mx, srow[j]);
      float sum = 0.f;
      for (int j = 0; j < 64; ++j) { float e_ = expf(srow[j]-mx); srow[j] = e_; sum += e_; }
      float inv = 1.f / sum;
      for (int j = 0; j < 64; ++j) srow[j] *= inv;
    }
    __syncthreads();

    // --- out_h = probs @ v_h (N padded 8->16; cols 8..15 discarded) ---
    for (int mt = 0; mt < 4; ++mt) {
      f32x8 c = {};
#pragma unroll
      for (int kt = 0; kt < 2; ++kt) {
        bf16x16 a = ld_frag_f32cvt(sc + (mt*16+lm)*64 + kt*32 + 8*lg);
        bf16x16 b = ld_frag_f32cvt(vT + (h*8+lm)*64 + kt*32 + 8*lg); // lm>=8: junk, discarded
        c = __builtin_amdgcn_wmma_f32_16x16x32_bf16(false, a, false, b, (short)0, c, false, false);
      }
      if (lm < 8) {
#pragma unroll
        for (int r = 0; r < 8; ++r)
          ao[(mt*16 + r + (lg << 3))*32 + h*8 + lm] = (__bf16)c[r];
      }
    }
    __syncthreads();
  }

  // --- proj: [64x32]@[32x64] + bias + shortcut -> xmid ---
  for (int i = lane; i < 64*32; i += 32) { int n = i >> 5, k = i & 31; wpT[i] = (__bf16)pw[k*64+n]; }
  __syncthreads();
  for (int mt = 0; mt < 4; ++mt)
    for (int nt = 0; nt < 4; ++nt) {
      bf16x16 a = ld_frag_bf16(ao  + (mt*16+lm)*32 + 8*lg);
      bf16x16 b = ld_frag_bf16(wpT + (nt*16+lm)*32 + 8*lg);
      f32x8 c = {};
      c = __builtin_amdgcn_wmma_f32_16x16x32_bf16(false, a, false, b, (short)0, c, false, false);
#pragma unroll
      for (int r = 0; r < 8; ++r) {
        int m = mt*16 + r + (lg << 3);
        int col = nt*16 + lm;
        long gt = bbase + (long)((wy*8 + (m >> 3))*W_ + (wx*8 + (m & 7)));
        xmid[gt*C_ + col] = x[gt*C_ + col] + c[r] + pbs[col];
      }
    }
}

// ---------------------------------------------------------------------------
// K3: LN2 + fc1 + exact GELU -> y (bf16). One wave = 64 tokens.
// ---------------------------------------------------------------------------
__global__ void __launch_bounds__(32)
k_ln2_fc1(const float* __restrict__ xmid,
          const float* __restrict__ g2, const float* __restrict__ b2,
          const float* __restrict__ w1, const float* __restrict__ b1,
          __bf16* __restrict__ ybuf) {
  __shared__ __align__(32) __bf16 xl[64*64];
  __shared__ __align__(32) __bf16 wfT[128*64];      // [n][k]
  __shared__ float sb[128], sg[64], sbe[64];
  int lane = threadIdx.x;
  long t0 = (long)blockIdx.x * 64;
  for (int i = lane; i < 128; i += 32) sb[i] = b1[i];
  for (int i = lane; i < 64;  i += 32) { sg[i] = g2[i]; sbe[i] = b2[i]; }
  for (int i = lane; i < 128*64; i += 32) { int n = i >> 6, k = i & 63; wfT[i] = (__bf16)w1[k*128+n]; }
  for (int rr = 0; rr < 2; ++rr) {
    int r = lane*2 + rr;
    const float* xr = xmid + (t0 + r)*C_;
    float v[64]; float mean = 0.f;
#pragma unroll
    for (int i = 0; i < 64; ++i) { v[i] = xr[i]; mean += v[i]; }
    mean *= (1.f/64.f);
    float var = 0.f;
#pragma unroll
    for (int i = 0; i < 64; ++i) { float d = v[i]-mean; var += d*d; }
    float rs = rsqrtf(var*(1.f/64.f) + 1e-5f);
#pragma unroll
    for (int i = 0; i < 64; ++i) xl[r*64+i] = (__bf16)((v[i]-mean)*rs*sg[i] + sbe[i]);
  }
  __syncthreads();
  int lm = lane & 15, lg = lane >> 4;
  for (int mt = 0; mt < 4; ++mt)
    for (int nt = 0; nt < 8; ++nt) {
      f32x8 c = {};
#pragma unroll
      for (int kt = 0; kt < 2; ++kt) {
        bf16x16 a = ld_frag_bf16(xl  + (mt*16+lm)*64 + kt*32 + 8*lg);
        bf16x16 b = ld_frag_bf16(wfT + (nt*16+lm)*64 + kt*32 + 8*lg);
        c = __builtin_amdgcn_wmma_f32_16x16x32_bf16(false, a, false, b, (short)0, c, false, false);
      }
#pragma unroll
      for (int r = 0; r < 8; ++r) {
        int m = mt*16 + r + (lg << 3);
        int col = nt*16 + lm;
        ybuf[(t0 + m)*HID_ + col] = (__bf16)gelu_exact(c[r] + sb[col]);
      }
    }
}

// ---------------------------------------------------------------------------
// K4: 5x5 depthwise conv + GELU + (y+yd)@fc2 + residual. Block = 8x8 tile.
// Halo tile loaded with async global->LDS copies.
// ---------------------------------------------------------------------------
__global__ void __launch_bounds__(128)
k_dw_fc2(const __bf16* __restrict__ ybuf,
         const float* __restrict__ dww, const float* __restrict__ dwb,
         const float* __restrict__ w2,  const float* __restrict__ b2,
         const float* __restrict__ xmid, float* __restrict__ out) {
  __shared__ __align__(32) __bf16 yt[12*12*128];    // 36864 B; z aliases after barrier
  __shared__ __align__(32) __bf16 wfT[64*128];      // [n][k] 16384 B
  __shared__ __bf16 dws[25*128];
  __shared__ float  sdb[128], sb2[64];
  __bf16* z = yt;                                   // [64][128]

  int tid = threadIdx.x;
  int tx = blockIdx.x, ty = blockIdx.y, batch = blockIdx.z;
  long bbase = (long)batch * NTOK;

  if (tid < 128) sdb[tid] = dwb[tid];
  if (tid < 64)  sb2[tid] = b2[tid];
  for (int i = tid; i < 25*128; i += 128) dws[i] = (__bf16)dww[i];
  for (int i = tid; i < 64*128; i += 128) { int n = i >> 7, k = i & 127; wfT[i] = (__bf16)w2[k*64+n]; }

  // pre-zero halo only on boundary blocks, then async-copy in-bounds 16B chunks
  bool edge = (tx == 0) | (ty == 0) | (tx == 31) | (ty == 31);
  if (edge) {
    for (int i = tid; i < 12*12*8; i += 128) ((uint4*)yt)[i] = make_uint4(0u,0u,0u,0u);
  }
  __syncthreads();
  unsigned ytb = lds_off(yt);
  for (int i = tid; i < 12*12*16; i += 128) {        // 16B = 8 channels per chunk
    int c8  = (i & 15) << 3;
    int pos = i >> 4;                                 // 0..143
    int ry = pos / 12, rx = pos % 12;
    int gy = ty*8 + ry - 2, gx = tx*8 + rx - 2;
    if (gy >= 0 && gy < H_ && gx >= 0 && gx < W_)
      async_b128(ytb + (unsigned)(pos*128 + c8)*2u,
                 ybuf + (bbase + (long)gy*W_ + gx)*HID_ + c8);
  }
  async_wait0();
  __syncthreads();

  // depthwise conv: thread -> token r = tid/2, 64 channels
  int r   = tid >> 1;
  int ch0 = (tid & 1) * 64;
  int py = r >> 3, px = r & 7;
  float acc[64];
#pragma unroll 1
  for (int c = 0; c < 64; ++c) {
    int ch = ch0 + c;
    float s = sdb[ch];
#pragma unroll
    for (int ky = 0; ky < 5; ++ky)
#pragma unroll
      for (int kx = 0; kx < 5; ++kx)
        s += (float)yt[((py+ky)*12 + (px+kx))*128 + ch] * (float)dws[(ky*5+kx)*128 + ch];
    float yc = (float)yt[((py+2)*12 + (px+2))*128 + ch];
    acc[c] = yc + gelu_exact(s);
  }
  __syncthreads();
  for (int c = 0; c < 64; ++c) z[r*128 + ch0 + c] = (__bf16)acc[c];
  __syncthreads();

  // fc2: [64x128]@[128x64]; wave w owns 16-col strip
  int lane = tid & 31, nt = tid >> 5;
  int lm = lane & 15, lg = lane >> 4;
  for (int mt = 0; mt < 4; ++mt) {
    f32x8 cc = {};
#pragma unroll
    for (int kt = 0; kt < 4; ++kt) {
      bf16x16 a = ld_frag_bf16(z   + (mt*16+lm)*128 + kt*32 + 8*lg);
      bf16x16 b = ld_frag_bf16(wfT + (nt*16+lm)*128 + kt*32 + 8*lg);
      cc = __builtin_amdgcn_wmma_f32_16x16x32_bf16(false, a, false, b, (short)0, cc, false, false);
    }
#pragma unroll
    for (int rr = 0; rr < 8; ++rr) {
      int m = mt*16 + rr + (lg << 3);
      int col = nt*16 + lm;
      long gt = bbase + (long)(ty*8 + (m >> 3))*W_ + (tx*8 + (m & 7));
      out[gt*C_ + col] = xmid[gt*C_ + col] + cc[rr] + sb2[col];
    }
  }
}

// ---------------------------------------------------------------------------
extern "C" void kernel_launch(void* const* d_in, const int* in_sizes, int n_in,
                              void* d_out, int out_size, void* d_ws, size_t ws_size,
                              hipStream_t stream) {
  (void)in_sizes; (void)n_in; (void)out_size; (void)ws_size;
  const float* x    = (const float*)d_in[0];
  const float* n1g  = (const float*)d_in[1];
  const float* n1b  = (const float*)d_in[2];
  const float* qw   = (const float*)d_in[3];
  const float* qb   = (const float*)d_in[4];
  const float* kvw  = (const float*)d_in[5];
  const float* kvb  = (const float*)d_in[6];
  const float* pw   = (const float*)d_in[7];
  const float* pb   = (const float*)d_in[8];
  const float* rpb  = (const float*)d_in[9];
  const float* pw1  = (const float*)d_in[10];
  const float* pb1  = (const float*)d_in[11];
  const float* pw2  = (const float*)d_in[12];
  const float* pb2  = (const float*)d_in[13];
  const float* n2g  = (const float*)d_in[14];
  const float* n2b  = (const float*)d_in[15];
  const float* f1w  = (const float*)d_in[16];
  const float* f1b  = (const float*)d_in[17];
  const float* dww  = (const float*)d_in[18];
  const float* dwb  = (const float*)d_in[19];
  const float* f2w  = (const float*)d_in[20];
  const float* f2b  = (const float*)d_in[21];

  char* ws = (char*)d_ws;
  __bf16* xn   = (__bf16*)(ws);                 // 16,777,216 B
  int*    fidx = (int*)(ws + 16777216);         //    524,288 B
  float*  xmid = (float*)(ws + 17301504);       // 33,554,432 B
  __bf16* ybuf = (__bf16*)(ws + 50855936);      // 33,554,432 B
  float*  outp = (float*)d_out;

  k_ln1_pred<<<512, 256, 0, stream>>>(x, n1g, n1b, pw1, pb1, pw2, pb2, xn, fidx);
  k_attn<<<2048, 32, 0, stream>>>(x, xn, fidx, qw, qb, kvw, kvb, pw, pb, rpb, xmid);
  k_ln2_fc1<<<2048, 32, 0, stream>>>(xmid, n2g, n2b, f1w, f1b, ybuf);
  k_dw_fc2<<<dim3(32, 32, 2), 128, 0, stream>>>(ybuf, dww, dwb, f2w, f2b, xmid, outp);
}
